// ReLuPCA_15874199126567
// MI455X (gfx1250) — matrix operand order repaired and checked by reading, not compiled, not run on hardware
//
#include <hip/hip_runtime.h>
#include <hip/hip_bf16.h>
#include <math.h>

// ---- problem constants (x = [64, 512, 28, 28] fp32, actBitwidth = 4) ----
#define N_IMG 64
#define NCH   512
#define HW    784
#define NHW   50176            // 64*784
#define TOT   (NCH*NHW)        // 25690112
#define NBINS 15               // 2^4 - 1
#define SWEEPS 6
#define KSPLIT 8               // cov K-chunks (50176/8 = 6272, multiple of 32)

typedef __attribute__((ext_vector_type(16))) __bf16 v16bf;
typedef __attribute__((ext_vector_type(8)))  float  v8f;
typedef __attribute__((ext_vector_type(4)))  unsigned u32x4;
typedef __attribute__((ext_vector_type(8)))  int      i32x8;
typedef __attribute__((ext_vector_type(4)))  int      i32x4;

#define WMMA_BF16(a, b, c) \
  __builtin_amdgcn_wmma_f32_16x16x32_bf16(false, (a), false, (b), (short)0, (c), false, false)

// ---- Tensor Data Mover availability (probe-verified builtins; arity differs) --------
#if defined(__has_builtin)
#if __has_builtin(__builtin_amdgcn_tensor_load_to_lds) && \
    __has_builtin(__builtin_amdgcn_s_wait_tensorcnt)
#define HAVE_TDM 1
#endif
#endif

#ifdef HAVE_TDM
// DMA a 2D tile (32 rows x 64 bf16 cols, row stride NHW elements) from global
// memory into LDS, packed contiguously (row-major [k][64]) -- exactly the sB
// layout the B-fragment reader consumes.  D# fields per CDNA5 ISA §8.3/8.4.
__device__ __forceinline__ void tdm_load_tile_b(const __bf16* gptr, unsigned lds_off) {
  unsigned long long ga = (unsigned long long)(const void*)gptr;
  u32x4 g0;
  g0[0] = 1u;                                  // count=1 (valid user descriptor)
  g0[1] = lds_off;                             // lds_addr (bytes), bits 63:32
  g0[2] = (unsigned)(ga & 0xFFFFFFFFu);        // global_addr[31:0], bits 95:64
  g0[3] = (unsigned)((ga >> 32) & 0x01FFFFFFu) // global_addr[56:32], bits 120:96
        | (2u << 30);                          // type=2 ("image"), bits 127:126
  i32x8 g1;
  g1[0] = (int)(1u << 16);                     // wg_mask=0, data_size=1 (2 bytes)
  g1[1] = (int)(((unsigned)NHW & 0xFFFFu) << 16);        // tensor_dim0[15:0] @63:48
  g1[2] = (int)((((unsigned)NHW >> 16) & 0xFFFFu)        // tensor_dim0[31:16]
        | (((unsigned)NCH & 0xFFFFu) << 16));            // tensor_dim1[15:0]
  g1[3] = (int)(64u << 16);                    // tensor_dim1[31:16]=0, tile_dim0=64
  g1[4] = 32;                                  // tile_dim1=32, tile_dim2=0 (2D tile)
  g1[5] = NHW;                                 // tensor_dim0_stride[31:0]
  g1[6] = 0;                                   // stride[47:32], dim1_stride low
  g1[7] = 0;
  i32x4 z4 = {0, 0, 0, 0};                     // groups 2/3 unused for 2D tensors
#if defined(__clang_major__) && __clang_major__ >= 23
  i32x8 z8 = {0, 0, 0, 0, 0, 0, 0, 0};
  __builtin_amdgcn_tensor_load_to_lds(g0, g1, z4, z4, z8, 0);
#else
  __builtin_amdgcn_tensor_load_to_lds(g0, g1, z4, z4, 0);
#endif
}
#endif

// ================= WMMA fragment loaders (CDNA5 wave32 layouts) =================
// A-matrix 16x32 bf16 (this code's contiguous k order, same as B): lane = half*16 + m; element e holds
// K = e + 16*half.  Loaded as 8 dwords (K pairs are contiguous).
__device__ __forceinline__ v16bf load_a_frag(const __bf16* A, int lda, int m0,
                                             int k0, int lane) {
  int half = lane >> 4, m = lane & 15;
  union { v16bf v; unsigned u[8]; } f;
  const unsigned* p = reinterpret_cast<const unsigned*>(A + (size_t)(m0 + m) * lda + k0);
#pragma unroll
  for (int d = 0; d < 8; ++d) {
    int dw = d + 8 * half;                  // dword index: K pair base = 2*dw
    f.u[d] = p[dw];
  }
  return f.v;
}

// B-matrix 32x16 bf16: lane = half*16 + n; element e holds K = e + 16*half.
// Variant 1: B(k,n) = M[n*ldk + k]  (row-of-M is K-contiguous) -> 8 dword loads.
__device__ __forceinline__ v16bf load_b_frag_rowk(const __bf16* M, int ldk, int n0,
                                                  int k0, int lane) {
  int half = lane >> 4, n = lane & 15;
  union { v16bf v; unsigned u[8]; } f;
  const unsigned* p = reinterpret_cast<const unsigned*>(M + (size_t)(n0 + n) * ldk + k0);
#pragma unroll
  for (int d = 0; d < 8; ++d) f.u[d] = p[d + 8 * half];
  return f.v;
}

// Variant 2: from an LDS-staged tile sB[32 k][64 n].
__device__ __forceinline__ v16bf load_b_frag_lds64(const __bf16* sB, int nlocal, int half) {
  union { v16bf v; __bf16 h[16]; } f;
#pragma unroll
  for (int e = 0; e < 16; ++e) f.h[e] = sB[(e + 16 * half) * 64 + nlocal];
  return f.v;
}

// ================= stage 1: ReLU + channel means, centered bf16 copy =================
__global__ void kChanMean(const float* __restrict__ x, float* __restrict__ chanMean) {
  int c = blockIdx.x, tid = threadIdx.x;
  float s = 0.f;
  for (int i = tid; i < NHW; i += 256) {
    int n = i / HW, hw = i % HW;
    s += fmaxf(x[(size_t)n * (NCH * HW) + (size_t)c * HW + hw], 0.f);
  }
  __shared__ float red[256];
  red[tid] = s; __syncthreads();
  for (int o = 128; o > 0; o >>= 1) { if (tid < o) red[tid] += red[tid + o]; __syncthreads(); }
  if (!tid) chanMean[c] = red[0] / (float)NHW;
}

__global__ void kBuildImC(const float* __restrict__ x, const float* __restrict__ chanMean,
                          __bf16* __restrict__ imC) {
  int idx = blockIdx.x * 256 + threadIdx.x;
  int c = idx / NHW, rem = idx % NHW;
  int n = rem / HW, hw = rem % HW;
  float v = fmaxf(x[(size_t)n * (NCH * HW) + (size_t)c * HW + hw], 0.f) - chanMean[c];
  imC[idx] = (__bf16)v;
}

// ================= stage 2: cov = imC * imC^T / NHW  (WMMA, 2x2 register tile) ========
__global__ void kZeroG(float* __restrict__ G) {
  G[blockIdx.x * 256 + threadIdx.x] = 0.f;
}

// 256 blocks = 32 tile-blocks x 8 K-chunks; each wave owns a 32x32 output super-tile
// for one K-chunk and atomically accumulates into Gcol.
__global__ void kCov(const __bf16* __restrict__ imC, float* __restrict__ Gcol) {
  int lane = threadIdx.x & 31, wid = threadIdx.x >> 5;
  int kc = blockIdx.x >> 5;                       // K-chunk 0..7
  int tile = (blockIdx.x & 31) * 8 + wid;         // 0..255 super-tiles (16x16 grid)
  int tm = tile & 15, tn = tile >> 4;
  int m0 = tm * 32, n0 = tn * 32;
  int kbeg = kc * (NHW / KSPLIT), kend = kbeg + (NHW / KSPLIT);
  v8f acc00 = {}, acc01 = {}, acc10 = {}, acc11 = {};
  for (int k0 = kbeg; k0 < kend; k0 += 32) {
    if (k0 + 32 < kend) {
      __builtin_prefetch(imC + (size_t)(m0 + (lane & 15)) * NHW + k0 + 32, 0, 1);
      __builtin_prefetch(imC + (size_t)(n0 + (lane & 15)) * NHW + k0 + 32, 0, 1);
    }
    v16bf a0 = load_a_frag(imC, NHW, m0,      k0, lane);
    v16bf a1 = load_a_frag(imC, NHW, m0 + 16, k0, lane);
    v16bf b0 = load_b_frag_rowk(imC, NHW, n0,      k0, lane);
    v16bf b1 = load_b_frag_rowk(imC, NHW, n0 + 16, k0, lane);
    acc00 = WMMA_BF16(a0, b0, acc00);
    acc01 = WMMA_BF16(a0, b1, acc01);
    acc10 = WMMA_BF16(a1, b0, acc10);
    acc11 = WMMA_BF16(a1, b1, acc11);
  }
  int half = lane >> 4, n = lane & 15;
  const float inv = 1.0f / (float)NHW;
#pragma unroll
  for (int r = 0; r < 8; ++r) {
    int m = r + 8 * half;
    atomicAdd(&Gcol[(size_t)(n0 + n)      * NCH + (m0 + m)],      acc00[r] * inv);
    atomicAdd(&Gcol[(size_t)(n0 + 16 + n) * NCH + (m0 + m)],      acc01[r] * inv);
    atomicAdd(&Gcol[(size_t)(n0 + n)      * NCH + (m0 + 16 + m)], acc10[r] * inv);
    atomicAdd(&Gcol[(size_t)(n0 + 16 + n) * NCH + (m0 + 16 + m)], acc11[r] * inv);
  }
}

// ================= stage 3: one-sided Jacobi eigensolver (cols are contiguous) ========
__global__ void kInitV(float* __restrict__ V) {
  int idx = blockIdx.x * 256 + threadIdx.x;
  int p = idx / NCH, c = idx % NCH;
  V[idx] = (p == c) ? 1.f : 0.f;
}

__global__ void kJacobi(float* __restrict__ G, float* __restrict__ V, int r) {
  const int m = NCH - 1;                    // 511, round-robin circle method
  int k = blockIdx.x;                       // 256 disjoint pairs
  int p, q;
  if (k == 0) { p = m; q = r % m; }
  else        { p = (r + k) % m; q = (r + m - k) % m; }
  float* Gp = G + (size_t)p * NCH; float* Gq = G + (size_t)q * NCH;
  float* Vp = V + (size_t)p * NCH; float* Vq = V + (size_t)q * NCH;
  int tid = threadIdx.x;
  float app = 0.f, aqq = 0.f, apq = 0.f;
  for (int i = tid; i < NCH; i += 256) {
    float gp = Gp[i], gq = Gq[i];
    app += gp * gp; aqq += gq * gq; apq += gp * gq;
  }
  __shared__ float r0[256], r1[256], r2[256];
  r0[tid] = app; r1[tid] = aqq; r2[tid] = apq; __syncthreads();
  for (int o = 128; o > 0; o >>= 1) {
    if (tid < o) { r0[tid] += r0[tid + o]; r1[tid] += r1[tid + o]; r2[tid] += r2[tid + o]; }
    __syncthreads();
  }
  __shared__ float cs[2];
  if (!tid) {
    float a = r0[0], b = r1[0], g = r2[0];
    float c = 1.f, s = 0.f;
    if (fabsf(g) > 1e-18f) {
      float tau = (b - a) / (2.f * g);
      float t = (tau >= 0.f ? 1.f : -1.f) / (fabsf(tau) + sqrtf(1.f + tau * tau));
      c = rsqrtf(1.f + t * t); s = t * c;
    }
    cs[0] = c; cs[1] = s;
  }
  __syncthreads();
  float c = cs[0], s = cs[1];
  for (int i = tid; i < NCH; i += 256) {
    float gp = Gp[i], gq = Gq[i];
    Gp[i] = c * gp - s * gq; Gq[i] = s * gp + c * gq;
    float vp = Vp[i], vq = Vq[i];
    Vp[i] = c * vp - s * vq; Vq[i] = s * vp + c * vq;
  }
}

// U = V (orthonormal eigenbasis).  Ut[m,k] = U[k,m] = Vcol[m*NCH+k];
// Ubf[c,p] = U[c,p] = Vcol[p*NCH+c].
__global__ void kConvU(const float* __restrict__ V, __bf16* __restrict__ Ut,
                       __bf16* __restrict__ Ubf) {
  int idx = blockIdx.x * 256 + threadIdx.x;
  Ut[idx] = (__bf16)V[idx];
  int c = idx / NCH, p = idx % NCH;
  Ubf[idx] = (__bf16)V[(size_t)p * NCH + c];
}

// ================= stage 4: imProj = Ut @ imC  (WMMA, 1x4 tiles/wave, TDM/LDS B) ======
// Block tile 128(M) x 64(N); 8 waves, each wave: one A frag reused over 4 B frags.
__global__ void kGemmProj(const __bf16* __restrict__ Abf, const __bf16* __restrict__ Bbf,
                          float* __restrict__ Dout) {
  __shared__ __align__(16) __bf16 sB[32 * 64];
  int tid = threadIdx.x, lane = tid & 31, wid = tid >> 5;
  int bm = blockIdx.x & 3, bn = blockIdx.x >> 2;
  int m0 = bm * 128 + wid * 16;
  int nb = bn * 64;
  v8f acc[4] = {{}, {}, {}, {}};
  for (int k0 = 0; k0 < NCH; k0 += 32) {
#ifdef HAVE_TDM
    if (wid == 0) {   // wave-uniform branch: TDM issues once, ignores EXEC
      tdm_load_tile_b(Bbf + (size_t)k0 * NHW + nb, (unsigned)(uintptr_t)&sB[0]);
      __builtin_amdgcn_s_wait_tensorcnt((short)0);
    }
#else
    {
      int lk = tid >> 3, ln = (tid & 7) * 8;      // cooperative 32x64 stage (16B/thread)
      *reinterpret_cast<uint4*>(&sB[lk * 64 + ln]) =
          *reinterpret_cast<const uint4*>(Bbf + (size_t)(k0 + lk) * NHW + nb + ln);
      if (k0 + 32 < NCH)
        __builtin_prefetch(Bbf + (size_t)(k0 + 32 + lk) * NHW + nb + ln, 0, 1);
    }
#endif
    __syncthreads();
    v16bf a = load_a_frag(Abf, NCH, m0, k0, lane);
#pragma unroll
    for (int j = 0; j < 4; ++j) {
      v16bf b = load_b_frag_lds64(sB, j * 16 + (lane & 15), lane >> 4);
      acc[j] = WMMA_BF16(a, b, acc[j]);
    }
    __syncthreads();
  }
  int half = lane >> 4, n = lane & 15;
#pragma unroll
  for (int j = 0; j < 4; ++j) {
    int col = nb + j * 16 + n;
#pragma unroll
    for (int r = 0; r < 8; ++r)
      Dout[(size_t)(m0 + r + 8 * half) * NHW + col] = acc[j][r];
  }
}

// ================= stage 5: per-row stats, Laplace clamp Newton, quant params =========
__global__ void kRowStats(const float* __restrict__ P, float* __restrict__ rMax,
                          float* __restrict__ rMin, float* __restrict__ rAbs) {
  int row = blockIdx.x, tid = threadIdx.x;
  const float* pr = P + (size_t)row * NHW;
  float mx = -3.4e38f, mn = 3.4e38f, sa = 0.f;
  for (int i = tid; i < NHW; i += 256) {
    float v = pr[i];
    mx = fmaxf(mx, v); mn = fminf(mn, v); sa += fabsf(v);
  }
  __shared__ float s0[256], s1[256], s2[256];
  s0[tid] = mx; s1[tid] = mn; s2[tid] = sa; __syncthreads();
  for (int o = 128; o > 0; o >>= 1) {
    if (tid < o) {
      s0[tid] = fmaxf(s0[tid], s0[tid + o]);
      s1[tid] = fminf(s1[tid], s1[tid + o]);
      s2[tid] += s2[tid + o];
    }
    __syncthreads();
  }
  if (!tid) { rMax[row] = s0[0]; rMin[row] = s1[0]; rAbs[row] = s2[0]; }
}

__global__ void kChanParams(const float* __restrict__ rMax, const float* __restrict__ rMin,
                            const float* __restrict__ rAbs, float* __restrict__ cvA,
                            float* __restrict__ dminA, float* __restrict__ scaleA,
                            float* __restrict__ multA, float* __restrict__ addA,
                            float* __restrict__ sameA) {
  int c = blockIdx.x * 256 + threadIdx.x;
  if (c >= NCH) return;
  float b = rAbs[c] / (float)NHW;
  float bs = fmaxf(b, 1e-12f);
  float alpha = 8.f * bs;
  const float kk = 12.f * 256.f;               // 12 * 2^(2*4)
  for (int it = 0; it < 30; ++it) {
    float e = expf(-alpha / bs);
    float g = -bs * e + alpha / kk;
    float gp = e + 1.f / kk;
    alpha = alpha - g / gp;
  }
  float cv = (b > 0.f) ? alpha : 0.f;
  float dmx = fminf(rMax[c], cv);              // range after clipping
  float dmn = fmaxf(rMin[c], -cv);
  int same = (dmx == dmn);
  float scale = same ? 1.f : 14.f / (dmx - dmn);   // 2^4 - 2
  cvA[c] = cv; dminA[c] = dmn; scaleA[c] = scale;
  multA[c] = same ? 1.f : 1.f / scale;
  addA[c]  = same ? 0.f : dmn;
  sameA[c] = (float)same;
}

// ================= stage 6: quantize + dequant (bf16) + histogram + row sums ==========
__global__ void kZeroHist(float* __restrict__ hist) {
  if (threadIdx.x < 16) hist[threadIdx.x] = 0.f;
}

__global__ void kQuant(const float* __restrict__ P, const float* __restrict__ cvA,
                       const float* __restrict__ dminA, const float* __restrict__ scaleA,
                       const float* __restrict__ multA, const float* __restrict__ addA,
                       const float* __restrict__ sameA, __bf16* __restrict__ Q,
                       float* __restrict__ rowQsum, float* __restrict__ hist) {
  int row = blockIdx.x, tid = threadIdx.x;
  __shared__ float lh[NBINS];
  __shared__ float red[256];
  if (tid < NBINS) lh[tid] = 0.f;
  __syncthreads();
  float cv = cvA[row], dmn = dminA[row], sc = scaleA[row];
  float mu = multA[row], ad = addA[row];
  bool same = sameA[row] != 0.f;
  const float* pr = P + (size_t)row * NHW;
  __bf16* qr = Q + (size_t)row * NHW;
  float qs = 0.f;
  for (int i = tid; i < NHW; i += 256) {
    float v = pr[i];
    v = fminf(fmaxf(v, -cv), cv);
    float q   = same ? v : rintf((v - dmn) * sc);   // round-to-nearest-even (RNE)
    float deq = same ? v : (q * mu + ad);
    int b = (int)q;
    if (b >= 0 && b < NBINS) atomicAdd(&lh[b], 1.f);  // OOB indices dropped (jax .at)
    qs += deq;
    qr[i] = (__bf16)deq;
  }
  red[tid] = qs; __syncthreads();
  for (int o = 128; o > 0; o >>= 1) { if (tid < o) red[tid] += red[tid + o]; __syncthreads(); }
  if (!tid) rowQsum[row] = red[0];
  if (tid < NBINS) atomicAdd(&hist[tid], lh[tid]);
}

// meanAdj[c] = mean over n of (U @ Q)[c,:] = sum_p U[c,p]*rowQsum[p] / NHW
__global__ void kMeanAdj(const float* __restrict__ V, const float* __restrict__ rowQsum,
                         float* __restrict__ meanAdj) {
  int c = blockIdx.x * 256 + threadIdx.x;
  if (c >= NCH) return;
  float s = 0.f;
  for (int p = 0; p < NCH; ++p) s += V[(size_t)p * NCH + c] * rowQsum[p];
  meanAdj[c] = s / (float)NHW;
}

__global__ void kEntropy(const float* __restrict__ hist, float* __restrict__ outp) {
  if (threadIdx.x == 0) {
    double tot = (double)TOT;
    double ent = 0.0;
    for (int i = 0; i < NBINS; ++i) {
      double p = (double)hist[i] / tot;
      if (p > 0.0) ent -= p * log2(p);
    }
    outp[0] = (float)ent;
    outp[1] = (float)(ent * tot);
  }
}

// ================= stage 7: out = U @ Qdeq - meanAdj + chanMean, scatter NCHW =========
__global__ void kGemmOut(const __bf16* __restrict__ Abf, const __bf16* __restrict__ Bbf,
                         const float* __restrict__ meanAdj, const float* __restrict__ chanMean,
                         float* __restrict__ out) {
  __shared__ __align__(16) __bf16 sB[32 * 64];
  int tid = threadIdx.x, lane = tid & 31, wid = tid >> 5;
  int bm = blockIdx.x & 3, bn = blockIdx.x >> 2;
  int m0 = bm * 128 + wid * 16;
  int nb = bn * 64;
  v8f acc[4] = {{}, {}, {}, {}};
  for (int k0 = 0; k0 < NCH; k0 += 32) {
#ifdef HAVE_TDM
    if (wid == 0) {
      tdm_load_tile_b(Bbf + (size_t)k0 * NHW + nb, (unsigned)(uintptr_t)&sB[0]);
      __builtin_amdgcn_s_wait_tensorcnt((short)0);
    }
#else
    {
      int lk = tid >> 3, ln = (tid & 7) * 8;
      *reinterpret_cast<uint4*>(&sB[lk * 64 + ln]) =
          *reinterpret_cast<const uint4*>(Bbf + (size_t)(k0 + lk) * NHW + nb + ln);
      if (k0 + 32 < NCH)
        __builtin_prefetch(Bbf + (size_t)(k0 + 32 + lk) * NHW + nb + ln, 0, 1);
    }
#endif
    __syncthreads();
    v16bf a = load_a_frag(Abf, NCH, m0, k0, lane);
#pragma unroll
    for (int j = 0; j < 4; ++j) {
      v16bf b = load_b_frag_lds64(sB, j * 16 + (lane & 15), lane >> 4);
      acc[j] = WMMA_BF16(a, b, acc[j]);
    }
    __syncthreads();
  }
  int half = lane >> 4, n = lane & 15;
  float adj[8];
#pragma unroll
  for (int r = 0; r < 8; ++r) {
    int m = m0 + r + 8 * half;
    adj[r] = chanMean[m] - meanAdj[m];
  }
#pragma unroll
  for (int j = 0; j < 4; ++j) {
    int col = nb + j * 16 + n;               // flat nhw index
    int nimg = col / HW, hw = col % HW;
#pragma unroll
    for (int r = 0; r < 8; ++r) {
      int m = m0 + r + 8 * half;             // channel
      out[(size_t)nimg * (NCH * HW) + (size_t)m * HW + hw] = acc[j][r] + adj[r];
    }
  }
}

// ======================================================================================
extern "C" void kernel_launch(void* const* d_in, const int* in_sizes, int n_in,
                              void* d_out, int out_size, void* d_ws, size_t ws_size,
                              hipStream_t stream) {
  (void)in_sizes; (void)n_in; (void)out_size; (void)ws_size;
  const float* x = (const float*)d_in[0];       // [64,512,28,28] fp32
  // d_in[1] = actBitwidth (4), baked into constants above
  float* out = (float*)d_out;                   // [out(TOT), bit_per_entry, bit_count]

  char* w = (char*)d_ws;
  size_t off = 0;
  auto carve = [&](size_t bytes) -> char* {
    char* p = w + off; off += (bytes + 255) & ~(size_t)255; return p;
  };
  __bf16* imC  = (__bf16*)carve((size_t)TOT * 2);        // centered activations, bf16
  float*  imP  = (float*) carve((size_t)TOT * 4);        // projected activations, fp32
  __bf16* Qdeq = (__bf16*)carve((size_t)TOT * 2);        // dequantized projection, bf16
  float*  G    = (float*) carve((size_t)NCH * NCH * 4);  // Jacobi working matrix (col-major)
  float*  V    = (float*) carve((size_t)NCH * NCH * 4);  // accumulated rotations = U (col-major)
  __bf16* Ut   = (__bf16*)carve((size_t)NCH * NCH * 2);  // U^T, row-major bf16
  __bf16* Ubf  = (__bf16*)carve((size_t)NCH * NCH * 2);  // U, row-major bf16
  float* chanMean = (float*)carve(NCH * 4);
  float* rMax  = (float*)carve(NCH * 4);
  float* rMin  = (float*)carve(NCH * 4);
  float* rAbs  = (float*)carve(NCH * 4);
  float* cvA   = (float*)carve(NCH * 4);
  float* dminA = (float*)carve(NCH * 4);
  float* scaleA= (float*)carve(NCH * 4);
  float* multA = (float*)carve(NCH * 4);
  float* addA  = (float*)carve(NCH * 4);
  float* sameA = (float*)carve(NCH * 4);
  float* rowQsum = (float*)carve(NCH * 4);
  float* meanAdj = (float*)carve(NCH * 4);
  float* hist  = (float*)carve(64);

  kChanMean<<<NCH, 256, 0, stream>>>(x, chanMean);
  kBuildImC<<<TOT / 256, 256, 0, stream>>>(x, chanMean, imC);
  kZeroG<<<NCH * NCH / 256, 256, 0, stream>>>(G);
  kCov<<<32 * KSPLIT, 256, 0, stream>>>(imC, G);
  kInitV<<<NCH * NCH / 256, 256, 0, stream>>>(V);
  for (int sweep = 0; sweep < SWEEPS; ++sweep)
    for (int r = 0; r < NCH - 1; ++r)
      kJacobi<<<NCH / 2, 256, 0, stream>>>(G, V, r);
  kConvU<<<NCH * NCH / 256, 256, 0, stream>>>(V, Ut, Ubf);
  kGemmProj<<<(NCH / 128) * (NHW / 64), 256, 0, stream>>>(Ut, imC, imP);
  kRowStats<<<NCH, 256, 0, stream>>>(imP, rMax, rMin, rAbs);
  kChanParams<<<2, 256, 0, stream>>>(rMax, rMin, rAbs, cvA, dminA, scaleA, multA, addA, sameA);
  kZeroHist<<<1, 32, 0, stream>>>(hist);
  kQuant<<<NCH, 256, 0, stream>>>(imP, cvA, dminA, scaleA, multA, addA, sameA,
                                  Qdeq, rowQsum, hist);
  kMeanAdj<<<2, 256, 0, stream>>>(V, rowQsum, meanAdj);
  kEntropy<<<1, 32, 0, stream>>>(hist, out + (size_t)TOT);
  kGemmOut<<<(NCH / 128) * (NHW / 64), 256, 0, stream>>>(Ubf, Qdeq, meanAdj, chanMean, out);
}
